// OSRA_80745385165174
// MI455X (gfx1250) — matrix operand order repaired and checked by reading, not compiled, not run on hardware
//
#include <hip/hip_runtime.h>
#include <hip/hip_bf16.h>

typedef _Float16 half16 __attribute__((ext_vector_type(16)));
typedef float    float8 __attribute__((ext_vector_type(8)));

#define DEV __device__ __forceinline__

// ---- WMMA fragment loaders (CDNA5 16x16x32 f16 layouts, wave32) ----

// A-matrix 16x32 (MxK): lane m=lane&15 (both halves), hi=lane>>4 selects
// K-range +16. rowp must already include + hi*16.
DEV half16 a_frag_from_row(const _Float16* rowp) {
  half16 a;
#pragma unroll
  for (int p = 0; p < 8; ++p) {
    int k = 2 * p; // {0,2,4,...,14}
    a[2 * p]     = rowp[k];
    a[2 * p + 1] = rowp[k + 1];
  }
  return a;
}

// B-matrix 32x16 (KxN): lane holds column n=lane&15, K = 16*(lane>>4)+e.
// From row-major M[n][k] the 16 K-values are contiguous.
DEV half16 b_frag_contig(const _Float16* colp) {
  half16 b;
#pragma unroll
  for (int e = 0; e < 16; ++e) b[e] = colp[e];
  return b;
}

// B[k][n] from row-major M[k][n]: stride ld between K values (LDS source).
DEV half16 b_frag_strided(const _Float16* basep, int ld) {
  half16 b;
#pragma unroll
  for (int e = 0; e < 16; ++e) b[e] = basep[e * ld];
  return b;
}

DEV float8 wmma_f16(half16 a, half16 b, float8 c) {
  return __builtin_amdgcn_wmma_f32_16x16x32_f16(false, a, false, b,
                                                (short)0, c, false, false);
}

// gfx1250 async global->LDS copy (16B per lane) + wait
DEV void async_copy_b128(unsigned lds_off, const void* gaddr) {
  asm volatile("global_load_async_to_lds_b128 %0, %1, off"
               :: "v"(lds_off), "v"((unsigned long long)(uintptr_t)gaddr)
               : "memory");
}
DEV void async_wait0() {
  asm volatile("s_wait_asynccnt 0x0" ::: "memory");
}

// ---- small utility kernels ----

__global__ void cvt_kernel(const float* __restrict__ src,
                           _Float16* __restrict__ dst, int n) {
  int i = blockIdx.x * blockDim.x + threadIdx.x;
  if (i < n) dst[i] = (_Float16)src[i];
}

// x [b,c,N] f32 -> xf [b,N,c] f16
__global__ void xf_kernel(const float* __restrict__ X, _Float16* __restrict__ XF) {
  int idx = blockIdx.x * blockDim.x + threadIdx.x;
  if (idx >= 16 * 1024 * 256) return;
  int cc = idx & 255;
  int n  = (idx >> 8) & 1023;
  int bb = idx >> 18;
  XF[idx] = (_Float16)X[((size_t)bb * 256 + cc) * 1024 + n];
}

// s = dwconv3x3(y) + dw_b + y   (all [b,N,c] f16, weights f32)
__global__ void dwconv_add_kernel(const _Float16* __restrict__ Y,
                                  const float* __restrict__ Wd,
                                  const float* __restrict__ Bd,
                                  _Float16* __restrict__ S) {
  int idx = blockIdx.x * blockDim.x + threadIdx.x;
  if (idx >= 16 * 256 * 1024) return;
  int n  = idx & 1023;
  int cc = (idx >> 10) & 255;
  int bb = idx >> 18;
  int i = n >> 5, j = n & 31;
  float acc = Bd[cc];
#pragma unroll
  for (int di = 0; di < 3; ++di) {
    int ii = i + di - 1;
    if (ii < 0 || ii > 31) continue;
#pragma unroll
    for (int dj = 0; dj < 3; ++dj) {
      int jj = j + dj - 1;
      if (jj < 0 || jj > 31) continue;
      acc += (float)Y[((size_t)(bb * 1024) + (ii << 5) + jj) * 256 + cc] *
             Wd[cc * 9 + di * 3 + dj];
    }
  }
  acc += (float)Y[((size_t)(bb * 1024) + n) * 256 + cc];
  S[((size_t)(bb * 1024) + n) * 256 + cc] = (_Float16)acc;
}

// ---- WMMA GEMM, 32x64 tile per wave: Out[r,o] = sum_k A[r,k]*W[o,k] (+bias) ----
__global__ __launch_bounds__(32) void gemm_wmma_kernel(
    const _Float16* __restrict__ A, const _Float16* __restrict__ W,
    const float* __restrict__ bias, _Float16* __restrict__ Out,
    int K, int O) {
  int lane = threadIdx.x & 31;
  int m  = lane & 15;
  int hi = lane >> 4;
  int r0 = blockIdx.y << 5;
  int o0 = blockIdx.x << 6;
  float8 acc[2][4];
#pragma unroll
  for (int rt = 0; rt < 2; ++rt)
#pragma unroll
    for (int ot = 0; ot < 4; ++ot)
#pragma unroll
      for (int j = 0; j < 8; ++j) acc[rt][ot][j] = 0.f;

  const _Float16* ar0 = A + (size_t)(r0 + m) * K + hi * 16;
  const _Float16* ar1 = ar0 + (size_t)16 * K;
  const _Float16* wr  = W + (size_t)(o0 + m) * K + hi * 16;
  for (int kt = 0; kt < K; kt += 32) {
    half16 a0 = a_frag_from_row(ar0 + kt);
    half16 a1 = a_frag_from_row(ar1 + kt);
    __builtin_prefetch(ar0 + kt + 32, 0, 0);
    __builtin_prefetch(ar1 + kt + 32, 0, 0);
#pragma unroll
    for (int ot = 0; ot < 4; ++ot) {
      half16 b = b_frag_contig(wr + (size_t)ot * 16 * K + kt);
      acc[0][ot] = wmma_f16(a0, b, acc[0][ot]);
      acc[1][ot] = wmma_f16(a1, b, acc[1][ot]);
    }
  }
  float bv[4];
#pragma unroll
  for (int ot = 0; ot < 4; ++ot) bv[ot] = bias ? bias[o0 + ot * 16 + m] : 0.f;
#pragma unroll
  for (int rt = 0; rt < 2; ++rt)
#pragma unroll
    for (int ot = 0; ot < 4; ++ot)
#pragma unroll
      for (int j = 0; j < 8; ++j)
        Out[(size_t)(r0 + rt * 16 + hi * 8 + j) * O + o0 + ot * 16 + m] =
            (_Float16)(acc[rt][ot][j] + bv[ot]);
}

// ---- fused flash attention, stage 1 (32 q-rows per wave) ----
// Q [b,N,256] (head hh cols hh*64..), KV [b,N,512] (k: ch, v: 256+ch)
__global__ __launch_bounds__(32) void attn1_kernel(
    const _Float16* __restrict__ Q, const _Float16* __restrict__ KV,
    _Float16* __restrict__ Yout) {
  __shared__ _Float16 ldsP[32 * 32];
  __shared__ _Float16 ldsV[32 * 64];
  int lane = threadIdx.x & 31;
  int m  = lane & 15;
  int hi = lane >> 4;
  int r0 = blockIdx.x << 5;
  int hh = blockIdx.y;
  int bb = blockIdx.z;

  half16 aq[2][2];
#pragma unroll
  for (int rt = 0; rt < 2; ++rt) {
    const _Float16* qrow =
        Q + ((size_t)(bb * 1024 + r0 + rt * 16 + m) * 256) + hh * 64 + hi * 16;
    aq[rt][0] = a_frag_from_row(qrow);
    aq[rt][1] = a_frag_from_row(qrow + 32);
  }

  float mrun[2][8], lrun[2][8];
  float8 o[2][4];
#pragma unroll
  for (int rt = 0; rt < 2; ++rt) {
#pragma unroll
    for (int j = 0; j < 8; ++j) { mrun[rt][j] = -1e30f; lrun[rt][j] = 0.f; }
#pragma unroll
    for (int dt = 0; dt < 4; ++dt)
#pragma unroll
      for (int j = 0; j < 8; ++j) o[rt][dt][j] = 0.f;
  }

  // per-lane async staging addresses: lane covers row (ch*4 + lane/8),
  // bytes (lane&7)*16 of the 128B (64 f16) V row
  unsigned ldsVoff = (unsigned)(uintptr_t)ldsV + (lane >> 3) * 128 + (lane & 7) * 16;
  const char* gV = (const char*)(KV + (size_t)(bb * 1024) * 512 + 256 + hh * 64) +
                   (size_t)(lane >> 3) * 1024 + (lane & 7) * 16;

  for (int k0 = 0; k0 < 1024; k0 += 32) {
    // kick off async V-tile copy (overlaps with QK^T + softmax below)
    const char* g = gV + (size_t)k0 * 1024;
#pragma unroll
    for (int ch = 0; ch < 8; ++ch)
      async_copy_b128(ldsVoff + ch * 512, g + ch * 4096);

    float8 s[2][2];
#pragma unroll
    for (int cb = 0; cb < 2; ++cb) {
      const _Float16* kcol =
          KV + ((size_t)(bb * 1024 + k0 + cb * 16 + m) * 512) + hh * 64 + hi * 16;
      half16 bk0 = b_frag_contig(kcol);
      half16 bk1 = b_frag_contig(kcol + 32);
#pragma unroll
      for (int rt = 0; rt < 2; ++rt) {
        float8 z;
#pragma unroll
        for (int j = 0; j < 8; ++j) z[j] = 0.f;
        z = wmma_f16(aq[rt][0], bk0, z);
        z = wmma_f16(aq[rt][1], bk1, z);
        s[rt][cb] = z;
      }
    }
#pragma unroll
    for (int rt = 0; rt < 2; ++rt) {
      float esc[8];
#pragma unroll
      for (int j = 0; j < 8; ++j) {
        float v0 = s[rt][0][j] * 0.125f;
        float v1 = s[rt][1][j] * 0.125f;
        float r = fmaxf(v0, v1);
        r = fmaxf(r, __shfl_xor(r, 1, 32));
        r = fmaxf(r, __shfl_xor(r, 2, 32));
        r = fmaxf(r, __shfl_xor(r, 4, 32));
        r = fmaxf(r, __shfl_xor(r, 8, 32));
        float mnew = fmaxf(mrun[rt][j], r);
        esc[j] = __expf(mrun[rt][j] - mnew);
        mrun[rt][j] = mnew;
        float p0 = __expf(v0 - mnew);
        float p1 = __expf(v1 - mnew);
        s[rt][0][j] = p0; s[rt][1][j] = p1;
        float rs = p0 + p1;
        rs += __shfl_xor(rs, 1, 32);
        rs += __shfl_xor(rs, 2, 32);
        rs += __shfl_xor(rs, 4, 32);
        rs += __shfl_xor(rs, 8, 32);
        lrun[rt][j] = lrun[rt][j] * esc[j] + rs;
      }
#pragma unroll
      for (int dt = 0; dt < 4; ++dt)
#pragma unroll
        for (int j = 0; j < 8; ++j) o[rt][dt][j] *= esc[j];
#pragma unroll
      for (int j = 0; j < 8; ++j) {
        int r = rt * 16 + hi * 8 + j;
        ldsP[r * 32 + m]      = (_Float16)s[rt][0][j];
        ldsP[r * 32 + 16 + m] = (_Float16)s[rt][1][j];
      }
    }
    async_wait0();
    __syncthreads();
    half16 ap[2];
#pragma unroll
    for (int rt = 0; rt < 2; ++rt)
      ap[rt] = a_frag_from_row(ldsP + (rt * 16 + m) * 32 + hi * 16);
#pragma unroll
    for (int dt = 0; dt < 4; ++dt) {
      half16 bv = b_frag_strided(ldsV + (hi * 16) * 64 + dt * 16 + m, 64);
#pragma unroll
      for (int rt = 0; rt < 2; ++rt)
        o[rt][dt] = wmma_f16(ap[rt], bv, o[rt][dt]);
    }
    __syncthreads();
  }

#pragma unroll
  for (int rt = 0; rt < 2; ++rt) {
#pragma unroll
    for (int j = 0; j < 8; ++j) lrun[rt][j] = 1.0f / lrun[rt][j];
#pragma unroll
    for (int dt = 0; dt < 4; ++dt)
#pragma unroll
      for (int j = 0; j < 8; ++j)
        Yout[(size_t)(bb * 1024 + r0 + rt * 16 + hi * 8 + j) * 256 +
             hh * 64 + dt * 16 + m] = (_Float16)(o[rt][dt][j] * lrun[rt][j]);
  }
}

// ---- fused flash attention, stage 2 (quirky reshape + scattered f32 out) ----
__global__ __launch_bounds__(32) void attn2_kernel(
    const _Float16* __restrict__ Q2, const _Float16* __restrict__ KV2,
    const float* __restrict__ rel_bias, float* __restrict__ Out) {
  __shared__ _Float16 ldsP[32 * 32];
  __shared__ _Float16 ldsV[32 * 64];
  int lane = threadIdx.x & 31;
  int m  = lane & 15;
  int hi = lane >> 4;
  int q0 = blockIdx.x << 5;   // tile over the 512 q-rows
  int hh = blockIdx.y;
  int tz = blockIdx.z;
  int t  = tz >> 4;
  int bb = tz & 15;

  // q2[t,b,h,u,d] = Q2[b, 2u+t, h*64+d]
  half16 aq[2][2];
#pragma unroll
  for (int rt = 0; rt < 2; ++rt) {
    const _Float16* qrow =
        Q2 + ((size_t)(bb * 1024 + 2 * (q0 + rt * 16 + m) + t) * 256) +
        hh * 64 + hi * 16;
    aq[rt][0] = a_frag_from_row(qrow);
    aq[rt][1] = a_frag_from_row(qrow + 32);
  }
  float rb = rel_bias[hh];

  float mrun[2][8], lrun[2][8];
  float8 o[2][4];
#pragma unroll
  for (int rt = 0; rt < 2; ++rt) {
#pragma unroll
    for (int j = 0; j < 8; ++j) { mrun[rt][j] = -1e30f; lrun[rt][j] = 0.f; }
#pragma unroll
    for (int dt = 0; dt < 4; ++dt)
#pragma unroll
      for (int j = 0; j < 8; ++j) o[rt][dt][j] = 0.f;
  }

  unsigned ldsVoff = (unsigned)(uintptr_t)ldsV + (lane >> 3) * 128 + (lane & 7) * 16;
  const char* gV = (const char*)(KV2 + (size_t)(bb * 1024) * 512 + 256 + hh * 64) +
                   (size_t)(lane >> 3) * 1024 + (lane & 7) * 16;

  for (int k0 = 0; k0 < 1024; k0 += 32) {
    const char* g = gV + (size_t)k0 * 1024;
#pragma unroll
    for (int ch = 0; ch < 8; ++ch)
      async_copy_b128(ldsVoff + ch * 512, g + ch * 4096);

    float8 s[2][2];
#pragma unroll
    for (int cb = 0; cb < 2; ++cb) {
      const _Float16* kcol =
          KV2 + ((size_t)(bb * 1024 + k0 + cb * 16 + m) * 512) + hh * 64 + hi * 16;
      half16 bk0 = b_frag_contig(kcol);
      half16 bk1 = b_frag_contig(kcol + 32);
#pragma unroll
      for (int rt = 0; rt < 2; ++rt) {
        float8 z;
#pragma unroll
        for (int j = 0; j < 8; ++j) z[j] = 0.f;
        z = wmma_f16(aq[rt][0], bk0, z);
        z = wmma_f16(aq[rt][1], bk1, z);
        s[rt][cb] = z;
      }
    }
#pragma unroll
    for (int rt = 0; rt < 2; ++rt) {
      float esc[8];
#pragma unroll
      for (int j = 0; j < 8; ++j) {
        float v0 = s[rt][0][j] * 0.125f + rb;
        float v1 = s[rt][1][j] * 0.125f + rb;
        float r = fmaxf(v0, v1);
        r = fmaxf(r, __shfl_xor(r, 1, 32));
        r = fmaxf(r, __shfl_xor(r, 2, 32));
        r = fmaxf(r, __shfl_xor(r, 4, 32));
        r = fmaxf(r, __shfl_xor(r, 8, 32));
        float mnew = fmaxf(mrun[rt][j], r);
        esc[j] = __expf(mrun[rt][j] - mnew);
        mrun[rt][j] = mnew;
        float p0 = __expf(v0 - mnew);
        float p1 = __expf(v1 - mnew);
        s[rt][0][j] = p0; s[rt][1][j] = p1;
        float rs = p0 + p1;
        rs += __shfl_xor(rs, 1, 32);
        rs += __shfl_xor(rs, 2, 32);
        rs += __shfl_xor(rs, 4, 32);
        rs += __shfl_xor(rs, 8, 32);
        lrun[rt][j] = lrun[rt][j] * esc[j] + rs;
      }
#pragma unroll
      for (int dt = 0; dt < 4; ++dt)
#pragma unroll
        for (int j = 0; j < 8; ++j) o[rt][dt][j] *= esc[j];
#pragma unroll
      for (int j = 0; j < 8; ++j) {
        int r = rt * 16 + hi * 8 + j;
        ldsP[r * 32 + m]      = (_Float16)s[rt][0][j];
        ldsP[r * 32 + 16 + m] = (_Float16)s[rt][1][j];
      }
    }
    async_wait0();
    __syncthreads();
    half16 ap[2];
#pragma unroll
    for (int rt = 0; rt < 2; ++rt)
      ap[rt] = a_frag_from_row(ldsP + (rt * 16 + m) * 32 + hi * 16);
#pragma unroll
    for (int dt = 0; dt < 4; ++dt) {
      half16 bv = b_frag_strided(ldsV + (hi * 16) * 64 + dt * 16 + m, 64);
#pragma unroll
      for (int rt = 0; rt < 2; ++rt)
        o[rt][dt] = wmma_f16(ap[rt], bv, o[rt][dt]);
    }
    __syncthreads();
  }

  // out[t,b,h,qq,d] -> Out[b2,c2,n2]: b2=8t+2h+(bb>>3), c2=(qq&3)*64+d,
  // n2=(bb&7)*128+(qq>>2)
  int b2 = t * 8 + hh * 2 + (bb >> 3);
#pragma unroll
  for (int rt = 0; rt < 2; ++rt) {
#pragma unroll
    for (int j = 0; j < 8; ++j) lrun[rt][j] = 1.0f / lrun[rt][j];
#pragma unroll
    for (int dt = 0; dt < 4; ++dt) {
      int d = dt * 16 + m;
#pragma unroll
      for (int j = 0; j < 8; ++j) {
        int qq = q0 + rt * 16 + hi * 8 + j;
        int c2 = (qq & 3) * 64 + d;
        int n2 = (bb & 7) * 128 + (qq >> 2);
        Out[((size_t)b2 * 256 + c2) * 1024 + n2] = o[rt][dt][j] * lrun[rt][j];
      }
    }
  }
}

extern "C" void kernel_launch(void* const* d_in, const int* in_sizes, int n_in,
                              void* d_out, int out_size, void* d_ws, size_t ws_size,
                              hipStream_t stream) {
  (void)in_sizes; (void)n_in; (void)out_size; (void)ws_size;
  const float* x      = (const float*)d_in[0];
  const float* q_w    = (const float*)d_in[1];
  const float* kv_w   = (const float*)d_in[2];
  const float* proj_w = (const float*)d_in[3];
  const float* proj_b = (const float*)d_in[4];
  const float* lq_w   = (const float*)d_in[5];
  const float* lq_b   = (const float*)d_in[6];
  const float* lkv_w  = (const float*)d_in[7];
  const float* lkv_b  = (const float*)d_in[8];
  const float* relb   = (const float*)d_in[9];
  const float* dw_w   = (const float*)d_in[10];
  const float* dw_b   = (const float*)d_in[11];

  char* ws = (char*)d_ws;
  const size_t MB = 1ull << 20;
  _Float16* wq   = (_Float16*)ws;          // 65536
  _Float16* wkv  = wq  + 65536;            // 131072
  _Float16* wpj  = wkv + 131072;           // 65536
  _Float16* wlq  = wpj + 65536;            // 65536
  _Float16* wlkv = wlq + 65536;            // 131072 (ends < 1MB)
  _Float16* xf   = (_Float16*)(ws + 1  * MB);  // [1,9)   live whole run
  _Float16* qh   = (_Float16*)(ws + 9  * MB);  // [9,17)  dead after attn1
  _Float16* kvh  = (_Float16*)(ws + 17 * MB);  // [17,33) dead after attn1
  _Float16* yat  = (_Float16*)(ws + 33 * MB);  // [33,41) dead after proj
  _Float16* yh   = (_Float16*)(ws + 9  * MB);  // reuse qh
  _Float16* sh   = (_Float16*)(ws + 17 * MB);  // reuse kvh lo
  _Float16* q2h  = (_Float16*)(ws + 25 * MB);  // reuse kvh hi
  _Float16* kv2h = (_Float16*)(ws + 33 * MB);  // [33,49) reuse yat + tail

  dim3 b256(256);
  cvt_kernel<<<(65536 + 255) / 256,  b256, 0, stream>>>(q_w,    wq,   65536);
  cvt_kernel<<<(131072 + 255) / 256, b256, 0, stream>>>(kv_w,   wkv,  131072);
  cvt_kernel<<<(65536 + 255) / 256,  b256, 0, stream>>>(proj_w, wpj,  65536);
  cvt_kernel<<<(65536 + 255) / 256,  b256, 0, stream>>>(lq_w,   wlq,  65536);
  cvt_kernel<<<(131072 + 255) / 256, b256, 0, stream>>>(lkv_w,  wlkv, 131072);
  xf_kernel<<<4194304 / 256, b256, 0, stream>>>(x, xf);

  // 32x64 tiles: grid = (O/64, R/32)
  gemm_wmma_kernel<<<dim3(4, 512), dim3(32), 0, stream>>>(xf,  wq,   nullptr, qh,   256, 256);
  gemm_wmma_kernel<<<dim3(8, 512), dim3(32), 0, stream>>>(xf,  wkv,  nullptr, kvh,  256, 512);
  attn1_kernel<<<dim3(32, 4, 16), dim3(32), 0, stream>>>(qh, kvh, yat);
  gemm_wmma_kernel<<<dim3(4, 512), dim3(32), 0, stream>>>(yat, wpj,  proj_b,  yh,   256, 256);
  dwconv_add_kernel<<<4194304 / 256, b256, 0, stream>>>(yh, dw_w, dw_b, sh);
  gemm_wmma_kernel<<<dim3(4, 512), dim3(32), 0, stream>>>(xf,  wlq,  lq_b,    q2h,  256, 256);
  gemm_wmma_kernel<<<dim3(8, 512), dim3(32), 0, stream>>>(sh,  wlkv, lkv_b,   kv2h, 256, 512);
  attn2_kernel<<<dim3(16, 4, 32), dim3(32), 0, stream>>>(q2h, kv2h, relb, (float*)d_out);
}
